// Grid_45105746542679
// MI455X (gfx1250) — matrix operand order repaired and checked by reading, not compiled, not run on hardware
//
#include <hip/hip_runtime.h>

// Fused instant-NGP hashgrid encode + 32->64->8 MLP (WMMA f16) + trilinear blend.
// MI455X / gfx1250, wave32. Table (42MB) lives in L2 (192MB); the only HBM
// traffic is xyz in, [N,8] out, and table first-touch. All intermediates stay
// in registers/LDS.

typedef __attribute__((ext_vector_type(16))) _Float16 v16h;
typedef __attribute__((ext_vector_type(8)))  float    v8f;

// Level tables replicated from the reference's _levels() (align_corners=False,
// base 16, desired 513, 16 levels, log2_hash 19, round-to-8).
static constexpr int      LRES[16] = {16,21,26,33,41,51,65,81,102,129,162,204,257,324,408,513};
static constexpr unsigned LHSH[16] = {4920u,10648u,19688u,39304u,74088u,140608u,287496u,
                                      524288u,524288u,524288u,524288u,524288u,524288u,524288u,524288u};
static constexpr int      LOFF[16] = {0,4920,15568,35256,74560,148648,289256,576752,
                                      1101040,1625328,2149616,2673904,3198192,3722480,4246768,4771056};
// levels 0..6 are dense (linear index % H), 7..15 hashed (& (2^19 - 1)).

template<int L0>
__device__ __forceinline__ void encode8(float ex, float ey, float ez,
                                        const float2* __restrict__ tb,
                                        _Float16* out16) {
#pragma unroll
  for (int j = 0; j < 8; ++j) {
    const int      lvl = L0 + j;                  // compile-time after unroll
    const float    R   = (float)LRES[lvl];
    const unsigned H   = LHSH[lvl];
    const int      O   = LOFF[lvl];
    float px = ex * R + 0.5f, py = ey * R + 0.5f, pz = ez * R + 0.5f;
    float fxf = floorf(px), fyf = floorf(py), fzf = floorf(pz);
    int   gx = (int)fxf, gy = (int)fyf, gz = (int)fzf;
    float fx = px - fxf, fy = py - fyf, fz = pz - fzf;
    float a0 = 0.f, a1 = 0.f;
#pragma unroll
    for (int k = 0; k < 8; ++k) {
      const int cx = (k >> 2) & 1, cy = (k >> 1) & 1, cz = k & 1;
      unsigned idx;
      if (lvl <= 6) {
        const unsigned R1 = (unsigned)(LRES[lvl] + 1);
        unsigned lin = (unsigned)(gx + cx)
                     + (unsigned)(gy + cy) * R1
                     + (unsigned)(gz + cz) * R1 * R1;
        idx = lin % H;                            // H is a constant -> mul/shift
      } else {
        idx = ((unsigned)(gx + cx) * 1u
             ^ (unsigned)(gy + cy) * 2654435761u
             ^ (unsigned)(gz + cz) * 805459861u) & (H - 1u);
      }
      float w = (cx ? fx : 1.f - fx) * (cy ? fy : 1.f - fy) * (cz ? fz : 1.f - fz);
      float2 t = tb[(long)O + (long)idx];         // global_load_b64 gather (L2-resident)
      a0 = fmaf(w, t.x, a0);
      a1 = fmaf(w, t.y, a1);
    }
    out16[2*j]   = (_Float16)a0;
    out16[2*j+1] = (_Float16)a1;
  }
}

// Pack w1 [32,64] and zero-padded w2 [64,16] into the per-lane WMMA B layout:
// entry (tile, lane, k) = B[8*(lane>>4) + (k&7) + 16*(k>>3) + 32*tile_k][16*tile_n + (lane&15)],
// so the GEMM kernel reads 16 contiguous halves (32B) per lane.
__global__ void pack_weights(const float* __restrict__ w1, const float* __restrict__ w2,
                             _Float16* __restrict__ b1p, _Float16* __restrict__ b2p) {
  int t = blockIdx.x * 256 + threadIdx.x;
  if (t < 2048) {                                  // b1p: 4 N-tiles x 32 lanes x 16
    int j = t >> 9, lane = (t >> 4) & 31, k = t & 15;
    int K = 8 * (lane >> 4) + (k & 7) + 16 * (k >> 3);
    int Ncol = 16 * j + (lane & 15);
    b1p[t] = (_Float16)w1[K * 64 + Ncol];
  } else if (t < 3072) {                           // b2p: 2 K-tiles x 32 lanes x 16
    int u = t - 2048;
    int kk = u >> 9, lane = (u >> 4) & 31, k = u & 15;
    int K = kk * 32 + 8 * (lane >> 4) + (k & 7) + 16 * (k >> 3);
    int Ncol = lane & 15;
    b2p[u] = (_Float16)((Ncol < 8) ? w2[K * 8 + Ncol] : 0.f);
  }
}

__global__ __launch_bounds__(256) void
fused_hashgrid_mlp(const float* __restrict__ xyz, const float* __restrict__ bound,
                   const float* __restrict__ table,
                   const _Float16* __restrict__ b1p, const _Float16* __restrict__ b2p,
                   float* __restrict__ out, int Npts) {
  __shared__ _Float16 ldsFeat[8][16 * 32];   // per-wave 16x32 f16 features (8 KB)
  __shared__ _Float16 ldsHid [8][16 * 64];   // per-wave 16x64 f16 hidden  (16 KB)

  const int tid  = threadIdx.x;
  const int wave = tid >> 5;
  const int lane = tid & 31;
  const int r    = lane & 15;     // M row within tile / B column within N-tile
  const int sel  = lane >> 4;     // K-half selector in the A/B stripings
  const int tile = blockIdx.x * 8 + wave;

  const float b     = bound[0];
  const float inv2b = 1.f / (2.f * b);

  // ---- hash-grid encoding: this lane covers row (tile*16 + r), levels sel*8..sel*8+7
  const int row = tile * 16 + r;                 // corner-point index, layout n*8 + c
  const int n   = min(row >> 3, Npts - 1);
  const int c   = row & 7;
  float nx = (xyz[n * 3 + 0] + b) * inv2b * 512.f;
  float ny = (xyz[n * 3 + 1] + b) * inv2b * 512.f;
  float nz = (xyz[n * 3 + 2] + b) * inv2b * 512.f;
  int cxi = min(max((int)floorf(nx), 0), 511) + ((c >> 2) & 1);
  int cyi = min(max((int)floorf(ny), 0), 511) + ((c >> 1) & 1);
  int czi = min(max((int)floorf(nz), 0), 511) + (c & 1);
  const float ex = (float)cxi * (1.f / 512.f);
  const float ey = (float)cyi * (1.f / 512.f);
  const float ez = (float)czi * (1.f / 512.f);

  union { _Float16 h[16]; float4 f4[2]; } fb;
  if (sel == 0) encode8<0>(ex, ey, ez, (const float2*)table, fb.h);
  else          encode8<8>(ex, ey, ez, (const float2*)table, fb.h);

  // 16 halves -> feats row r, columns [16*sel, 16*sel+16): contiguous 32B in LDS
  {
    float4* dst = (float4*)&ldsFeat[wave][r * 32 + sel * 16];
    dst[0] = fb.f4[0];
    dst[1] = fb.f4[1];
  }
  __syncthreads();

  // ---- layer 1: A = feats 16x32 f16 in WMMA A striping
  v16h a1;
  {
    const _Float16* fr = &ldsFeat[wave][r * 32];
    ((float4*)&a1)[0] = *(const float4*)(fr + sel * 8);        // K = 8*sel .. +8
    ((float4*)&a1)[1] = *(const float4*)(fr + 16 + sel * 8);   // K = 16+8*sel .. +8
  }

#pragma unroll
  for (int j = 0; j < 4; ++j) {                  // 4 N-tiles of 16 -> hidden dim 64
    v16h bj = *(const v16h*)(b1p + (j * 32 + lane) * 16);
    v8f cj = {};
    cj = __builtin_amdgcn_wmma_f32_16x16x32_f16(false, a1, false, bj,
                                                (short)0, cj, false, false);
#pragma unroll
    for (int i = 0; i < 8; ++i) {                // ReLU -> f16 -> LDS row-major
      float v = cj[i];
      v = v > 0.f ? v : 0.f;
      ldsHid[wave][(i + 8 * sel) * 64 + (j * 16 + r)] = (_Float16)v;
    }
  }
  __syncthreads();

  // ---- layer 2: hidden 16x64 @ w2pad 64x16, K split into two chained WMMAs
  v8f d = {};
  const _Float16* hr = &ldsHid[wave][r * 64];
#pragma unroll
  for (int kk = 0; kk < 2; ++kk) {
    v16h a2;
    ((float4*)&a2)[0] = *(const float4*)(hr + kk * 32 + sel * 8);
    ((float4*)&a2)[1] = *(const float4*)(hr + kk * 32 + 16 + sel * 8);
    v16h b2 = *(const v16h*)(b2p + (kk * 32 + lane) * 16);
    d = __builtin_amdgcn_wmma_f32_16x16x32_f16(false, a2, false, b2,
                                               (short)0, d, false, false);
  }

  // ---- trilinear blend: this lane's 8 D values are the 8 corner results of
  // point np = tile*2 + sel at output channel r (valid for r < 8).
  const int np = min(tile * 2 + sel, Npts - 1);
  float qx = (xyz[np * 3 + 0] + b) * inv2b * 512.f;
  float qy = (xyz[np * 3 + 1] + b) * inv2b * 512.f;
  float qz = (xyz[np * 3 + 2] + b) * inv2b * 512.f;
  float u = qx - (float)min(max((int)floorf(qx), 0), 511);
  float v = qy - (float)min(max((int)floorf(qy), 0), 511);
  float w = qz - (float)min(max((int)floorf(qz), 0), 511);
  float o = 0.f;
#pragma unroll
  for (int k = 0; k < 8; ++k) {
    float wk = (((k >> 2) & 1) ? u : 1.f - u)
             * (((k >> 1) & 1) ? v : 1.f - v)
             * (((k     ) & 1) ? w : 1.f - w);
    o = fmaf(wk, d[k], o);
  }
  if (r < 8 && (tile * 2 + sel) < Npts) out[np * 8 + r] = o;
}

extern "C" void kernel_launch(void* const* d_in, const int* in_sizes, int n_in,
                              void* d_out, int out_size, void* d_ws, size_t ws_size,
                              hipStream_t stream) {
  const float* xyz   = (const float*)d_in[0];
  const float* bound = (const float*)d_in[1];
  const float* table = (const float*)d_in[2];
  const float* w1    = (const float*)d_in[3];
  const float* w2    = (const float*)d_in[4];
  float* out = (float*)d_out;
  const int N = in_sizes[0] / 3;                  // 262144

  _Float16* b1p = (_Float16*)d_ws;                // 4*32*16 = 2048 halves (4 KB)
  _Float16* b2p = b1p + 2048;                     // 2*32*16 = 1024 halves (2 KB)

  pack_weights<<<12, 256, 0, stream>>>(w1, w2, b1p, b2p);

  // 8N corner rows, 16 rows per wave-tile, 8 waves per block.
  const int tiles  = (N * 8 + 15) / 16;
  const int blocks = (tiles + 7) / 8;             // N=262144 -> exactly 16384
  fused_hashgrid_mlp<<<blocks, 256, 0, stream>>>(xyz, bound, table, b1p, b2p, out, N);
}
